// ProjectionLayer_34943853920384
// MI455X (gfx1250) — matrix operand order repaired and checked by reading, not compiled, not run on hardware
//
#include <hip/hip_runtime.h>

#define B_   8192
#define DIN  1024
#define UN   1024
#define EN   16
#define APAD 8                      // +16B row pad -> conflict-spread A-fragment ds_load_b128
#define ASTR (DIN + APAD)
#define MROWS 32                    // batch rows per block
#define NWAVE 16                    // waves per block (512 threads)

typedef __attribute__((ext_vector_type(16))) __bf16 bf16x16;
typedef __attribute__((ext_vector_type(8)))  float  f32x8;

union BFrag { bf16x16 v; uint4 q[2]; };

__device__ __forceinline__ unsigned short f2bf(float f) {
    unsigned int u = __float_as_uint(f);
    u += 0x7FFFu + ((u >> 16) & 1u);          // round-to-nearest-even
    return (unsigned short)(u >> 16);
}

// ---------------------------------------------------------------------------
// Kernel 1: W[E, D, U] f32  ->  Wt[E, U, D] bf16  (LDS-tiled transpose+convert)
// ---------------------------------------------------------------------------
__global__ __launch_bounds__(256)
void wt_transpose_kernel(const float* __restrict__ W, unsigned short* __restrict__ Wt) {
    __shared__ float tile[32][33];
    const int e  = blockIdx.z;
    const int u0 = blockIdx.x * 32;
    const int d0 = blockIdx.y * 32;
    const int tx = threadIdx.x;               // 0..31
    const int ty = threadIdx.y;               // 0..7
    const float* src = W + (size_t)e * DIN * UN;
    unsigned short* dst = Wt + (size_t)e * UN * DIN;
#pragma unroll
    for (int i = 0; i < 4; ++i) {
        int d = ty + i * 8;
        tile[d][tx] = src[(size_t)(d0 + d) * UN + (u0 + tx)];   // coalesced in u
    }
    __syncthreads();
#pragma unroll
    for (int i = 0; i < 4; ++i) {
        int u = ty + i * 8;
        dst[(size_t)(u0 + u) * DIN + (d0 + tx)] = f2bf(tile[tx][u]); // coalesced in d
    }
}

// ---------------------------------------------------------------------------
// Kernel 2: 32-row batch tile per block; 16 waves x (64 cols x 2 row-tiles).
// Every B fragment feeds 2 WMMAs -> 1 global b128 per WMMA (halved L2 traffic).
// ---------------------------------------------------------------------------
__global__ __launch_bounds__(512)
void proj_softmax_kernel(const float* __restrict__ x,
                         const float* __restrict__ tp,
                         const unsigned short* __restrict__ Wt,
                         const float* __restrict__ bias,
                         float* __restrict__ out) {
    __shared__ unsigned short Alds[MROWS][ASTR];  // x tile, bf16 bits (~66 KB)
    __shared__ float tp_lds[MROWS * EN];          // [row 32][entity 16]
    __shared__ float partred[NWAVE][MROWS];       // per-wave row partials
    __shared__ float scratch32[MROWS];            // rowmax / coeff broadcast

    const int tid     = threadIdx.x;              // 0..511
    const int wave    = tid >> 5;                 // 0..15
    const int lane    = tid & 31;
    const int nn      = lane & 15;                // tile column / A-row (M within tile)
    const int hi      = lane >> 4;                // half-wave id
    const int rowbase = blockIdx.x * MROWS;
    const int colbase = wave * 64;                // 4 col tiles per wave

    // ---- Phase 0: stage x tile f32 -> bf16 LDS, preload type_prob ----
#pragma unroll
    for (int it = 0; it < 16; ++it) {
        int fi = tid + it * 512;                  // float4 index, 0..8191
        int r  = fi >> 8;                         // row 0..31
        int kc = (fi & 255) << 2;
        const float4 v = *(const float4*)(x + (size_t)(rowbase + r) * DIN + kc);
        uint2 p;
        p.x = (unsigned)f2bf(v.x) | ((unsigned)f2bf(v.y) << 16);
        p.y = (unsigned)f2bf(v.z) | ((unsigned)f2bf(v.w) << 16);
        *(uint2*)&Alds[r][kc] = p;
    }
    tp_lds[tid] = tp[(size_t)rowbase * EN + tid]; // 512 values = 32 rows x 16 entities
    __syncthreads();

    f32x8 oacc[2][4];
#pragma unroll
    for (int rt = 0; rt < 2; ++rt)
#pragma unroll
        for (int ct = 0; ct < 4; ++ct)
#pragma unroll
            for (int c = 0; c < 8; ++c) oacc[rt][ct][c] = 0.0f;

    for (int e = 0; e < EN; ++e) {
        // ---- logits = x @ W[e] + b[e]  (bf16 WMMA, f32 accumulate) ----
        f32x8 acc[2][4];
#pragma unroll
        for (int ct = 0; ct < 4; ++ct) {
            float bv = bias[(size_t)e * UN + colbase + ct * 16 + nn];
#pragma unroll
            for (int c = 0; c < 8; ++c) { acc[0][ct][c] = bv; acc[1][ct][c] = bv; }
        }
        const unsigned short* wbase = Wt + (size_t)e * UN * DIN;
        for (int k0 = 0; k0 < DIN; k0 += 32) {
            BFrag a0, a1;                         // A fragments for the two row-tiles
            a0.q[0] = *(const uint4*)&Alds[nn][k0 + hi * 8];
            a0.q[1] = *(const uint4*)&Alds[nn][k0 + 16 + hi * 8];
            a1.q[0] = *(const uint4*)&Alds[nn + 16][k0 + hi * 8];
            a1.q[1] = *(const uint4*)&Alds[nn + 16][k0 + 16 + hi * 8];
#pragma unroll
            for (int ct = 0; ct < 4; ++ct) {
                const unsigned short* brow =
                    wbase + (size_t)(colbase + ct * 16 + nn) * DIN + k0 + hi * 8;
                BFrag b;
                b.q[0] = *(const uint4*)(brow);
                b.q[1] = *(const uint4*)(brow + 16);
                acc[0][ct] = __builtin_amdgcn_wmma_f32_16x16x32_bf16(
                    false, a0.v, false, b.v, (short)0, acc[0][ct], false, false);
                acc[1][ct] = __builtin_amdgcn_wmma_f32_16x16x32_bf16(
                    false, a1.v, false, b.v, (short)0, acc[1][ct], false, false);
            }
        }

        // ---- row max over U (lane: rows m = rt*16 + 8*hi + j) ----
        float mj[2][8];
#pragma unroll
        for (int rt = 0; rt < 2; ++rt)
#pragma unroll
            for (int j = 0; j < 8; ++j) {
                float m = acc[rt][0][j];
#pragma unroll
                for (int ct = 1; ct < 4; ++ct) m = fmaxf(m, acc[rt][ct][j]);
#pragma unroll
                for (int msk = 1; msk < 16; msk <<= 1)
                    m = fmaxf(m, __shfl_xor(m, msk, 32));
                mj[rt][j] = m;
            }
        if (nn == 0)
#pragma unroll
            for (int rt = 0; rt < 2; ++rt)
#pragma unroll
                for (int j = 0; j < 8; ++j)
                    partred[wave][rt * 16 + hi * 8 + j] = mj[rt][j];
        __syncthreads();
        if (tid < MROWS) {
            float m = partred[0][tid];
#pragma unroll
            for (int w = 1; w < NWAVE; ++w) m = fmaxf(m, partred[w][tid]);
            scratch32[tid] = m;
        }
        __syncthreads();
        float rmax[2][8];
#pragma unroll
        for (int rt = 0; rt < 2; ++rt)
#pragma unroll
            for (int j = 0; j < 8; ++j) rmax[rt][j] = scratch32[rt * 16 + hi * 8 + j];

        // ---- exp + row sum ----
        float sj[2][8];
#pragma unroll
        for (int rt = 0; rt < 2; ++rt)
#pragma unroll
            for (int j = 0; j < 8; ++j) sj[rt][j] = 0.0f;
#pragma unroll
        for (int rt = 0; rt < 2; ++rt)
#pragma unroll
            for (int ct = 0; ct < 4; ++ct)
#pragma unroll
                for (int j = 0; j < 8; ++j) {
                    float ev = __expf(acc[rt][ct][j] - rmax[rt][j]);
                    acc[rt][ct][j] = ev;
                    sj[rt][j] += ev;
                }
#pragma unroll
        for (int rt = 0; rt < 2; ++rt)
#pragma unroll
            for (int j = 0; j < 8; ++j)
#pragma unroll
                for (int msk = 1; msk < 16; msk <<= 1)
                    sj[rt][j] += __shfl_xor(sj[rt][j], msk, 32);
        if (nn == 0)
#pragma unroll
            for (int rt = 0; rt < 2; ++rt)
#pragma unroll
                for (int j = 0; j < 8; ++j)
                    partred[wave][rt * 16 + hi * 8 + j] = sj[rt][j];
        __syncthreads();
        if (tid < MROWS) {
            float s = partred[0][tid];
#pragma unroll
            for (int w = 1; w < NWAVE; ++w) s += partred[w][tid];
            scratch32[tid] = tp_lds[tid * EN + e] / s;     // tp[b,e] / rowsum
        }
        __syncthreads();
        float cj[2][8];
#pragma unroll
        for (int rt = 0; rt < 2; ++rt)
#pragma unroll
            for (int j = 0; j < 8; ++j) cj[rt][j] = scratch32[rt * 16 + hi * 8 + j];
#pragma unroll
        for (int rt = 0; rt < 2; ++rt)
#pragma unroll
            for (int ct = 0; ct < 4; ++ct)
#pragma unroll
                for (int j = 0; j < 8; ++j) oacc[rt][ct][j] += cj[rt][j] * acc[rt][ct][j];
    }

    // ---- store out[32 x 64] per wave region ----
#pragma unroll
    for (int rt = 0; rt < 2; ++rt)
#pragma unroll
        for (int ct = 0; ct < 4; ++ct)
#pragma unroll
            for (int j = 0; j < 8; ++j)
                out[(size_t)(rowbase + rt * 16 + hi * 8 + j) * UN + colbase + ct * 16 + nn] =
                    oacc[rt][ct][j];
}

// ---------------------------------------------------------------------------
extern "C" void kernel_launch(void* const* d_in, const int* in_sizes, int n_in,
                              void* d_out, int out_size, void* d_ws, size_t ws_size,
                              hipStream_t stream) {
    (void)in_sizes; (void)n_in; (void)out_size; (void)ws_size;
    const float* x    = (const float*)d_in[0];
    const float* tp   = (const float*)d_in[1];
    const float* W    = (const float*)d_in[2];
    const float* bias = (const float*)d_in[3];
    float* out        = (float*)d_out;
    unsigned short* Wt = (unsigned short*)d_ws;   // E*U*D bf16 = 32 MB scratch

    wt_transpose_kernel<<<dim3(UN / 32, DIN / 32, EN), dim3(32, 8), 0, stream>>>(W, Wt);
    proj_softmax_kernel<<<dim3(B_ / MROWS), dim3(NWAVE * 32), 0, stream>>>(x, tp, Wt, bias, out);
}
